// LinearAttention_19361712570558
// MI455X (gfx1250) — matrix operand order repaired and checked by reading, not compiled, not run on hardware
//
#include <hip/hip_runtime.h>

// LinearAttention fused block for MI455X (gfx1250), wave32 + WMMA bf16.
// f32 in HBM, bf16 tiles in LDS, v_wmma_f32_16x16x32_bf16, f32 accumulation.

#define DD   128   // d_model
#define TS   128   // S rows per block
#define NH   2     // heads
#define DKH  64    // head dim
#define BB   8     // batch
#define SS   8192  // sequence

typedef __attribute__((ext_vector_type(16))) __bf16 v16bf;
typedef __attribute__((ext_vector_type(8)))  __bf16 v8bf;
typedef __attribute__((ext_vector_type(4)))  __bf16 v4bf;
typedef __attribute__((ext_vector_type(8)))  float  v8f;

__device__ __forceinline__ v8f wmma_bf16(v16bf a, v16bf b, v8f c) {
  // (neg_a, A, neg_b, B, c_mod, C, reuse_a, reuse_b)
  return __builtin_amdgcn_wmma_f32_16x16x32_bf16(false, a, false, b, (short)0, c,
                                                 false, false);
}

// ---- fragment builders (contiguous k = 16h + s order for A and B, wave32) -------------------
// A (16x32 bf16): lane<16 -> M=lane, K = 0..15; lane>=16 -> K = 16..31.
// Two contiguous 16B runs per lane -> 2x ds_load_b128.
__device__ __forceinline__ v16bf load_a_rm(const __bf16* src, int pitch,
                                           int row0, int col0, int lane) {
  const int m = lane & 15;
  const int koff = (lane & 16) ? 16 : 0;
  const __bf16* p = src + (row0 + m) * pitch + col0 + koff;
  v8bf lo = *(const v8bf*)p;         // K = koff + 0..7
  v8bf hi = *(const v8bf*)(p + 8);   // K = koff + 8..15
  return __builtin_shufflevector(lo, hi, 0, 1, 2, 3, 4, 5, 6, 7,
                                 8, 9, 10, 11, 12, 13, 14, 15);
}
// transposed gather (KV accumulation): logical A[m][k] = src[(k0+k)*pitch + m0+m]
__device__ __forceinline__ v16bf load_a_tr(const __bf16* src, int pitch,
                                           int k0, int m0, int lane) {
  const int m = lane & 15;
  const int koff = (lane & 16) ? 16 : 0;
  v16bf a;
#pragma unroll
  for (int i = 0; i < 16; ++i) {
    const int k = i + koff;
    a[i] = src[(k0 + k) * pitch + m0 + m];
  }
  return a;
}
// B (32x16 bf16): lane n holds column n; lanes 0-15 K=0..15, lanes 16-31 K=16..31.
// Row-major source (column gather, strided): B[k][n] = src[(k0+k)*pitch + n0+n]
__device__ __forceinline__ v16bf load_b_rm(const __bf16* src, int pitch,
                                           int k0, int n0, int lane) {
  const int n = lane & 15;
  const int koff = (lane & 16) ? 16 : 0;
  const __bf16* p = src + (k0 + koff) * pitch + n0 + n;
  v16bf b;
#pragma unroll
  for (int i = 0; i < 16; ++i) b[i] = p[i * pitch];
  return b;
}
// B from a TRANSPOSED bf16 LDS matrix T (T[n][k] = B[k][n]): one contiguous
// 32-byte run per lane -> 2x ds_load_b128.
__device__ __forceinline__ v16bf load_b_tr(const __bf16* T, int pitch,
                                           int n0, int k0, int lane) {
  const int n = lane & 15;
  const int koff = (lane & 16) ? 16 : 0;
  const __bf16* p = T + (n0 + n) * pitch + k0 + koff;
  v8bf lo = *(const v8bf*)p;
  v8bf hi = *(const v8bf*)(p + 8);
  return __builtin_shufflevector(lo, hi, 0, 1, 2, 3, 4, 5, 6, 7,
                                 8, 9, 10, 11, 12, 13, 14, 15);
}
// B for x@W^T from row-major f32 W in global: B[k][n] = W[n0+n][k0+k];
// contiguous per lane, L2-resident, loop-invariant (compiler hoists).
__device__ __forceinline__ v16bf load_bW_g(const float* W, int n0, int k0, int lane) {
  const int n = lane & 15;
  const int koff = (lane & 16) ? 16 : 0;
  const float* p = W + (n0 + n) * DD + k0 + koff;
  v16bf b;
#pragma unroll
  for (int i = 0; i < 16; ++i) b[i] = (__bf16)p[i];
  return b;
}

// cooperative f32 global -> bf16 LDS tile copy (packed converts, b64 LDS stores)
__device__ __forceinline__ void tile_f32_to_bf16(__bf16* dst, const float* src,
                                                 int nelem, int tid) {
  for (int i = tid * 4; i < nelem; i += 256 * 4) {
    const float4 f = *(const float4*)(src + i);
    v4bf t;
    t[0] = (__bf16)f.x; t[1] = (__bf16)f.y; t[2] = (__bf16)f.z; t[3] = (__bf16)f.w;
    *(v4bf*)(dst + i) = t;
  }
}

// phi(x) = elu(x) + 1
__device__ __forceinline__ float phi(float x) {
  return (x > 0.f) ? (x + 1.f) : __expf(x);
}

// ===========================================================================
// Kernel 1: per (b, S-chunk): Kp = phi(k@Wk^T+bk), Vp = v@Wv^T+bv  (WMMA),
// then partial KV[h] += Kp_h^T @ Vp_h (WMMA) and Ksum[h] += colsum(Kp_h),
// accumulated into f32 workspace with hardware fp atomics.  LDS: 96 KB.
// ===========================================================================
__global__ __launch_bounds__(256) void kv_proj_accum(
    const float* __restrict__ kin, const float* __restrict__ vin,
    const float* __restrict__ Wk, const float* __restrict__ bk,
    const float* __restrict__ Wv, const float* __restrict__ bv,
    float* __restrict__ kvws, float* __restrict__ ksumws) {
  __shared__ __bf16 sIn[TS * DD];   // 32 KB, bf16 input tile
  __shared__ __bf16 sKp[TS * DD];   // 32 KB
  __shared__ __bf16 sVp[TS * DD];   // 32 KB

  const int tid = threadIdx.x, lane = tid & 31, wave = tid >> 5;
  const int b = blockIdx.x >> 6;
  const int chunk = blockIdx.x & 63;
  const size_t base = ((size_t)b * SS + (size_t)chunk * TS) * DD;

  for (int p = 0; p < 2; ++p) {
    const float* Wsrc = p ? Wv : Wk;
    const float* bsrc = p ? bv : bk;
    const float* xsrc = (p ? vin : kin) + base;
    __bf16* dst = p ? sVp : sKp;
    __syncthreads();  // previous pass done reading sIn before overwrite
    tile_f32_to_bf16(sIn, xsrc, TS * DD, tid);
    __syncthreads();
    // 64 output tiles of 16x16; wave w owns column strip nt=w, mt=0..7
    for (int t = wave; t < 64; t += 8) {
      const int mt = t >> 3, nt = t & 7;
      v8f acc = {};
#pragma unroll
      for (int kk = 0; kk < 4; ++kk) {
        v16bf a = load_a_rm(sIn, DD, mt * 16, kk * 32, lane);
        v16bf w = load_bW_g(Wsrc, nt * 16, kk * 32, lane);
        acc = wmma_bf16(a, w, acc);
      }
      const int n = lane & 15, mb = (lane & 16) ? 8 : 0;
      const float bias = bsrc[nt * 16 + n];
#pragma unroll
      for (int r = 0; r < 8; ++r) {
        float x = acc[r] + bias;
        if (p == 0) x = phi(x);          // phi only on K
        dst[(mt * 16 + mb + r) * DD + nt * 16 + n] = (__bf16)x;
      }
    }
  }
  __syncthreads();

  // partial KV[h][d][e] = sum_s Kp[s, h*64+d] * Vp[s, h*64+e]
  for (int t = wave; t < 32; t += 8) {
    const int h = t >> 4, tt = t & 15, mt = tt >> 2, nt = tt & 3;
    const int cb = h * DKH;
    v8f acc = {};
#pragma unroll
    for (int kk = 0; kk < 4; ++kk) {   // K dim = 128 chunk rows
      v16bf a = load_a_tr(sKp, DD, kk * 32, cb + mt * 16, lane);
      v16bf v = load_b_rm(sVp, DD, kk * 32, cb + nt * 16, lane);
      acc = wmma_bf16(a, v, acc);
    }
    const int n = lane & 15, mb = (lane & 16) ? 8 : 0;
    float* out = kvws + (size_t)(b * NH + h) * DKH * DKH;
#pragma unroll
    for (int r = 0; r < 8; ++r)
      unsafeAtomicAdd(&out[(mt * 16 + mb + r) * DKH + nt * 16 + n], acc[r]);
  }
  // partial Ksum: thread t<128 reduces column t of Kp
  if (tid < DD) {
    float s = 0.f;
    for (int r = 0; r < TS; ++r) s += (float)sKp[r * DD + tid];
    unsafeAtomicAdd(&ksumws[(size_t)b * DD + tid], s);
  }
}

// ===========================================================================
// Kernel 2: per (b, S-chunk): Qp = phi(q@Wq^T+bq); Z = 1/(Qp . Ksum);
// attn = (Qp @ KV) * Z; out = attn @ Wo^T + bo; x = out + rs*q; LayerNorm.
// sAux overlays the q-bf16 tile (GEMM1 input) and the attn tile (GEMM2
// output) -- separated by two barriers.  LDS: ~145 KB.
// ===========================================================================
__global__ __launch_bounds__(256) void attn_out(
    const float* __restrict__ q, const float* __restrict__ Wq,
    const float* __restrict__ bq, const float* __restrict__ Wo,
    const float* __restrict__ bo, const float* __restrict__ gamma,
    const float* __restrict__ beta, const float* __restrict__ res_scale,
    const float* __restrict__ kvws, const float* __restrict__ ksumws,
    float* __restrict__ out) {
  __shared__ float  sQ[TS * DD];           // 64 KB: q f32, later pre-LN x
  __shared__ __bf16 sAux[TS * DD];         // 32 KB: q bf16, later attn bf16
  __shared__ __bf16 sQp[TS * DD];          // 32 KB
  __shared__ __bf16 sKVT[NH * DKH * DKH];  // 16 KB, KV transposed: [h][e][d]
  __shared__ float  sKsum[DD];
  __shared__ float  sZ[TS * NH];

  const int tid = threadIdx.x, lane = tid & 31, wave = tid >> 5;
  const int b = blockIdx.x >> 6;
  const int chunk = blockIdx.x & 63;
  const size_t base = ((size_t)b * SS + (size_t)chunk * TS) * DD;

  // q tile: keep f32 (residual/LN) and bf16 (GEMM1 A) copies
  for (int i = tid * 4; i < TS * DD; i += 256 * 4) {
    const float4 f = *(const float4*)(q + base + i);
    *(float4*)(sQ + i) = f;
    v4bf t;
    t[0] = (__bf16)f.x; t[1] = (__bf16)f.y; t[2] = (__bf16)f.z; t[3] = (__bf16)f.w;
    *(v4bf*)(sAux + i) = t;
  }
  // KV, scattered transposed so GEMM2 B-fragments are contiguous
  {
    const size_t bofs = (size_t)b * NH * DKH * DKH;
    for (int i = tid; i < NH * DKH * DKH; i += 256) {
      const int h = i >> 12, d = (i >> 6) & 63, e = i & 63;
      sKVT[(h << 12) + (e << 6) + d] = (__bf16)kvws[bofs + i];
    }
  }
  if (tid < DD) sKsum[tid] = ksumws[(size_t)b * DD + tid];
  __syncthreads();

  // --- GEMM1: Qp = phi(q @ Wq^T + bq) ---
  for (int t = wave; t < 64; t += 8) {
    const int mt = t >> 3, nt = t & 7;
    v8f acc = {};
#pragma unroll
    for (int kk = 0; kk < 4; ++kk) {
      v16bf a = load_a_rm(sAux, DD, mt * 16, kk * 32, lane);
      v16bf w = load_bW_g(Wq, nt * 16, kk * 32, lane);
      acc = wmma_bf16(a, w, acc);
    }
    const int n = lane & 15, mb = (lane & 16) ? 8 : 0;
    const float bias = bq[nt * 16 + n];
#pragma unroll
    for (int r = 0; r < 8; ++r)
      sQp[(mt * 16 + mb + r) * DD + nt * 16 + n] = (__bf16)phi(acc[r] + bias);
  }
  __syncthreads();

  // --- Z[s,h] = 1 / (Qp[s, h*64: ] . Ksum[h]) ; (s,h) = (tid/2, tid&1)
  {
    const int s = tid >> 1, h = tid & 1;
    float d = 0.f;
#pragma unroll 8
    for (int i = 0; i < DKH; ++i)
      d += (float)sQp[s * DD + h * DKH + i] * sKsum[h * DKH + i];
    sZ[s * NH + h] = 1.f / d;
  }
  __syncthreads();

  // --- GEMM2: attn[s, h*64+e] = (sum_d Qp[s,h*64+d]*KV[h][d][e]) * Z[s,h]
  //     (writes into sAux; GEMM1 reads of sAux finished two barriers ago)
  for (int t = wave; t < 64; t += 8) {
    const int mt = t >> 3, nt = t & 7;
    const int h = nt >> 2, cb = h * DKH;
    v8f acc = {};
#pragma unroll
    for (int kk = 0; kk < 2; ++kk) {   // K = 64 (head dim)
      v16bf a = load_a_rm(sQp, DD, mt * 16, cb + kk * 32, lane);
      v16bf kv = load_b_tr(sKVT + h * DKH * DKH, DKH, (nt & 3) * 16, kk * 32, lane);
      acc = wmma_bf16(a, kv, acc);
    }
    const int n = lane & 15, mb = (lane & 16) ? 8 : 0;
#pragma unroll
    for (int r = 0; r < 8; ++r) {
      const int row = mt * 16 + mb + r;
      sAux[row * DD + nt * 16 + n] = (__bf16)(acc[r] * sZ[row * NH + h]);
    }
  }
  __syncthreads();

  // --- GEMM3: x = attn @ Wo^T + bo + rs*q  (write back into sQ, f32) ---
  const float rs = res_scale[0];
  for (int t = wave; t < 64; t += 8) {
    const int mt = t >> 3, nt = t & 7;
    v8f acc = {};
#pragma unroll
    for (int kk = 0; kk < 4; ++kk) {
      v16bf a = load_a_rm(sAux, DD, mt * 16, kk * 32, lane);
      v16bf w = load_bW_g(Wo, nt * 16, kk * 32, lane);
      acc = wmma_bf16(a, w, acc);
    }
    const int n = lane & 15, mb = (lane & 16) ? 8 : 0;
    const float bias = bo[nt * 16 + n];
#pragma unroll
    for (int r = 0; r < 8; ++r) {
      const int idx = (mt * 16 + mb + r) * DD + nt * 16 + n;
      sQ[idx] = acc[r] + bias + rs * sQ[idx];   // same-thread RMW, no race
    }
  }
  __syncthreads();

  // --- per-row LayerNorm (thread t < 128 owns row t) ---
  if (tid < TS) {
    const float* xr = sQ + tid * DD;
    float mu = 0.f;
    for (int j = 0; j < DD; ++j) mu += xr[j];
    mu *= (1.f / DD);
    float var = 0.f;
    for (int j = 0; j < DD; ++j) { float d = xr[j] - mu; var += d * d; }
    var *= (1.f / DD);
    const float rstd = rsqrtf(var + 1e-5f);
    float* orow = out + base + (size_t)tid * DD;
    for (int j = 0; j < DD; ++j)
      orow[j] = (xr[j] - mu) * rstd * gamma[j] + beta[j];
  }
}

// ===========================================================================
extern "C" void kernel_launch(void* const* d_in, const int* in_sizes, int n_in,
                              void* d_out, int out_size, void* d_ws, size_t ws_size,
                              hipStream_t stream) {
  const float* q   = (const float*)d_in[0];
  const float* k   = (const float*)d_in[1];
  const float* v   = (const float*)d_in[2];
  const float* Wq  = (const float*)d_in[3];
  const float* bq  = (const float*)d_in[4];
  const float* Wk  = (const float*)d_in[5];
  const float* bk  = (const float*)d_in[6];
  const float* Wv  = (const float*)d_in[7];
  const float* bv  = (const float*)d_in[8];
  const float* Wo  = (const float*)d_in[9];
  const float* bo  = (const float*)d_in[10];
  const float* gm  = (const float*)d_in[11];
  const float* bt  = (const float*)d_in[12];
  const float* rsc = (const float*)d_in[13];
  float* out = (float*)d_out;

  float* kvws   = (float*)d_ws;                       // [B][H][64][64] f32
  float* ksumws = kvws + (size_t)BB * NH * DKH * DKH; // [B][H][64]    f32
  const size_t ws_bytes =
      (size_t)(BB * NH * DKH * DKH + BB * NH * DKH) * sizeof(float);

  hipMemsetAsync(d_ws, 0, ws_bytes, stream);          // graph-capturable

  const dim3 blk(256);
  const dim3 grd(BB * (SS / TS));  // 512 blocks
  kv_proj_accum<<<grd, blk, 0, stream>>>(k, v, Wk, bk, Wv, bv, kvws, ksumws);
  attn_out<<<grd, blk, 0, stream>>>(q, Wq, bq, Wo, bo, gm, bt, rsc,
                                    kvws, ksumws, out);
}
